// Hgrn2_53884659695718
// MI455X (gfx1250) — matrix operand order repaired and checked by reading, not compiled, not run on hardware
//
#include <hip/hip_runtime.h>
#include <hip/hip_bf16.h>
#include <math.h>

// Problem constants (match reference)
#define B_   4
#define L_   8192
#define E_   1024
#define H_   64
#define D_   2
#define F_   128        // H*D
#define N3_  384        // 3*F (q|f|i fused projection width)
#define BL_  (B_*L_)    // 32768 rows
#define TC_  128        // scan chunk length
#define NC_  (L_/TC_)   // 64 chunks

typedef __attribute__((ext_vector_type(16))) __bf16 bf16x16;
typedef __attribute__((ext_vector_type(8)))  __bf16 bf16x8;
typedef __attribute__((ext_vector_type(8)))  float  fx8;

union FragU { bf16x16 v; bf16x8 h[2]; };

__device__ __forceinline__ bf16x16 make_frag(const __bf16* p0, const __bf16* p1) {
  FragU u;
  u.h[0] = *reinterpret_cast<const bf16x8*>(p0);
  u.h[1] = *reinterpret_cast<const bf16x8*>(p1);
  return u.v;
}

// ---------------------------------------------------------------------------
// Weight prep: fp32 -> bf16, transposed to N-major so B-fragments are
// two contiguous 8-element K-runs per lane, at k=8h and k=16+8h (matches V_WMMA B layout).
// ---------------------------------------------------------------------------
__global__ void prep_W(const float* __restrict__ Wq, const float* __restrict__ Wf,
                       const float* __restrict__ Wi, __bf16* __restrict__ WT) {
  int i = blockIdx.x * blockDim.x + threadIdx.x;
  if (i >= N3_ * E_) return;
  int n = i / E_, kk = i % E_;
  float w;
  if (n < F_)        w = Wq[(size_t)kk * F_ + n];
  else if (n < 2*F_) w = Wf[(size_t)kk * F_ + (n - F_)];
  else               w = Wi[(size_t)kk * F_ + (n - 2*F_)];
  WT[i] = (__bf16)w;
}

__global__ void prep_Wo(const float* __restrict__ Wo, __bf16* __restrict__ WoT) {
  int i = blockIdx.x * blockDim.x + threadIdx.x;
  if (i >= E_ * F_) return;
  int n = i / F_, kk = i % F_;
  WoT[i] = (__bf16)Wo[(size_t)kk * E_ + n];
}

// ---------------------------------------------------------------------------
// Fused projection GEMM: z[32768,384] = x[32768,1024] @ [Wq|Wf|Wi]
// Epilogue applies gelu / (1-sigmoid, log(sigmoid+eps)) / identity.
// Block: 256 threads = 8 waves; one 16-row M tile staged in LDS as bf16
// (fp32->bf16 conversion happens en route, so this stays a VGPR-mediated
// copy on purpose), each wave owns 3 N-tiles (8*3*16 = 384).
// ---------------------------------------------------------------------------
__global__ __launch_bounds__(256) void proj_gemm(
    const float* __restrict__ x, const __bf16* __restrict__ WT,
    float* __restrict__ qo, float* __restrict__ ko,
    float* __restrict__ vo, float* __restrict__ go) {
  __shared__ __align__(16) __bf16 As[16][1032];   // pad: 1032*2B=2064B => +4 banks/row
  const int tid = threadIdx.x;
  const int m0  = blockIdx.x * 16;

  // Stage 16x1024 fp32 rows of x as bf16 into LDS
  const float4* xv = reinterpret_cast<const float4*>(x);
  for (int i = tid; i < 16 * 256; i += 256) {
    int row = i >> 8, c4 = i & 255;
    float4 f = xv[(size_t)(m0 + row) * 256 + c4];
    int c = c4 * 4;
    As[row][c+0] = (__bf16)f.x;  As[row][c+1] = (__bf16)f.y;
    As[row][c+2] = (__bf16)f.z;  As[row][c+3] = (__bf16)f.w;
  }
  __syncthreads();

  const int wave = tid >> 5, lane = tid & 31;
  const int m = lane & 15, grp = lane >> 4;

  fx8 acc[3] = {};
  for (int kb = 0; kb < E_; kb += 32) {
    bf16x16 a = make_frag(&As[m][kb + grp*8], &As[m][kb + 16 + grp*8]);
#pragma unroll
    for (int j = 0; j < 3; ++j) {
      int nt = wave*3 + j;
      const __bf16* bp = WT + (size_t)(nt*16 + m) * E_ + kb + grp*8;
      bf16x16 bfr = make_frag(bp, bp + 16);
      acc[j] = __builtin_amdgcn_wmma_f32_16x16x32_bf16(
          false, a, false, bfr, (short)0, acc[j], false, false);
    }
  }

#pragma unroll
  for (int j = 0; j < 3; ++j) {
    int nt = wave*3 + j;
    int region = nt >> 3;               // 16-wide tiles never straddle 128-col regions
    int colin  = (nt*16 + m) & 127;
#pragma unroll
    for (int r = 0; r < 8; ++r) {
      int row = m0 + grp*8 + r;
      size_t idx = (size_t)row * F_ + colin;
      float val = acc[j][r];
      if (region == 0) {                 // q = gelu(z) (exact erf form)
        qo[idx] = 0.5f * val * (1.0f + erff(val * 0.70710678118654752f));
      } else if (region == 1) {          // f -> g=sigmoid; k=1-g; gl=log(g+1e-6)
        float s = 1.0f / (1.0f + expf(-val));
        ko[idx] = 1.0f - s;
        go[idx] = logf(s + 1e-6f);
      } else {                           // v = i
        vo[idx] = val;
      }
    }
  }
}

// ---------------------------------------------------------------------------
// Chunked GLA scan, pass 1: per (b, chunk) block, 128 threads = (h, dk).
// Local scan from S=0; emits local outputs, decayed-q (aq = q*scale*exp(cumg)),
// and per-chunk summaries (final local state + total log-decay).
// ---------------------------------------------------------------------------
__global__ __launch_bounds__(128) void scan_local(
    const float* __restrict__ q, const float* __restrict__ k,
    const float* __restrict__ v, const float* __restrict__ gl,
    float* __restrict__ oloc, float* __restrict__ aq,
    float* __restrict__ Send, float* __restrict__ Gtot) {
  const int b = blockIdx.x / NC_;
  const int c = blockIdx.x % NC_;
  const int j = threadIdx.x;            // h*2 + dk
  const int dk = j & 1;
  const float scale = 0.70710678118654752f;   // D**-0.5

  float S0 = 0.f, S1 = 0.f, cum = 0.f;
  size_t base = ((size_t)b * L_ + (size_t)c * TC_) * F_ + j;

  for (int i = 0; i < TC_; ++i) {
    size_t idx = base + (size_t)i * F_;
    if (i + 1 < TC_) {
      __builtin_prefetch(&k[idx + F_], 0, 0);
      __builtin_prefetch(&gl[idx + F_], 0, 0);
    }
    float kk = k[idx];
    float g  = gl[idx];
    float qv = q[idx] * scale;
    float vv = v[idx];
    float vn = __shfl_xor(vv, 1);       // pair lane holds the other dv channel
    float v0 = dk ? vn : vv;
    float v1 = dk ? vv : vn;
    float e  = expf(g);
    S0 = e * S0 + kk * v0;              // S[dk][dv] = exp(g[dk])*S + k[dk]*v[dv]
    S1 = e * S1 + kk * v1;
    cum += g;
    aq[idx] = qv * expf(cum);           // q_t * exp(prefix log-decay incl. t)
    float x0 = qv * S0, x1 = qv * S1;   // this dk's contribution to o[dv]
    float o0 = x0 + __shfl_xor(x0, 1);
    float o1 = x1 + __shfl_xor(x1, 1);
    oloc[idx] = dk ? o1 : o0;           // lane dk writes o[dv=dk] (same channel j)
  }

  const int h = j >> 1;
  size_t soff = ((((size_t)b * H_ + h) * NC_ + c) * 2 + dk) * 2;
  Send[soff + 0] = S0;
  Send[soff + 1] = S1;
  Gtot[(((size_t)b * H_ + h) * NC_ + c) * 2 + dk] = cum;
}

// Pass 2: serial combine across 64 chunks; 512 independent (b,h,dk) state rows.
__global__ __launch_bounds__(512) void scan_combine(
    const float* __restrict__ Send, const float* __restrict__ Gtot,
    float* __restrict__ Sinit) {
  int t = threadIdx.x;
  int b = t >> 7, h = (t >> 1) & 63, dk = t & 1;
  float s0 = 0.f, s1 = 0.f;
  for (int c = 0; c < NC_; ++c) {
    size_t off = ((((size_t)b * H_ + h) * NC_ + c) * 2 + dk) * 2;
    Sinit[off + 0] = s0;                // state entering chunk c
    Sinit[off + 1] = s1;
    float e = expf(Gtot[(((size_t)b * H_ + h) * NC_ + c) * 2 + dk]);
    s0 = e * s0 + Send[off + 0];
    s1 = e * s1 + Send[off + 1];
  }
}

// Pass 3: correction o_t += aq_t[dk] dot Sinit[dk][dv]; then tanh -> bf16.
__global__ __launch_bounds__(128) void scan_fix(
    const float* __restrict__ oloc, const float* __restrict__ aq,
    const float* __restrict__ Sinit, __bf16* __restrict__ obf) {
  const int b = blockIdx.x / NC_;
  const int c = blockIdx.x % NC_;
  const int j = threadIdx.x;            // h*2 + dv
  const int h = j >> 1, d = j & 1;

  size_t sbase = (((size_t)b * H_ + h) * NC_ + c) * 4;   // [dk][dv]
  float s0d = Sinit[sbase + 0 + d];
  float s1d = Sinit[sbase + 2 + d];

  size_t base = ((size_t)b * L_ + (size_t)c * TC_) * F_;
  for (int i = 0; i < TC_; ++i) {
    size_t rb = base + (size_t)i * F_;
    if (i + 1 < TC_) __builtin_prefetch(&oloc[rb + F_ + j], 0, 0);
    float a0 = aq[rb + h*2 + 0];
    float a1 = aq[rb + h*2 + 1];
    float val = oloc[rb + j] + a0 * s0d + a1 * s1d;
    obf[rb + j] = (__bf16)tanhf(val);
  }
}

// ---------------------------------------------------------------------------
// Output GEMM: out[32768,1024] = tanh_o_bf16[32768,128] @ Wo[128,1024] (fp32 out)
// Block: 8 waves, 16-row M tile staged into LDS via GLOBAL_LOAD_ASYNC_TO_LDS
// (A is already bf16 in global memory -> no conversion needed, so the async
// LDS-direct path applies: ASYNCcnt-tracked, no VGPR round trip).
// Each wave owns 8 N-tiles.
// ---------------------------------------------------------------------------
__global__ __launch_bounds__(256) void out_gemm(
    const __bf16* __restrict__ obf, const __bf16* __restrict__ WoT,
    float* __restrict__ out) {
  __shared__ __align__(16) __bf16 As[16][136];    // 272B stride => +4 banks/row
  const int tid = threadIdx.x;
  const int m0  = blockIdx.x * 16;

  {
    // One 16B chunk per thread: 16 rows x 128 bf16 = 256 chunks.
    int row = tid >> 4, c8 = tid & 15;
    unsigned int lds_off = (unsigned int)(row * 136 + c8 * 8) * 2u;  // byte addr in LDS segment
    const __bf16* gp = obf + (size_t)(m0 + row) * F_ + c8 * 8;
    asm volatile("global_load_async_to_lds_b128 %0, %1, off"
                 :: "v"(lds_off), "v"(gp)
                 : "memory");
    asm volatile("s_wait_asynccnt 0x0" ::: "memory");
  }
  __syncthreads();

  const int wave = tid >> 5, lane = tid & 31;
  const int m = lane & 15, grp = lane >> 4;

  fx8 acc[8] = {};
  for (int kb = 0; kb < F_; kb += 32) {
    bf16x16 a = make_frag(&As[m][kb + grp*8], &As[m][kb + 16 + grp*8]);
#pragma unroll
    for (int j = 0; j < 8; ++j) {
      int n = (wave*8 + j)*16 + m;
      const __bf16* bp = WoT + (size_t)n * F_ + kb + grp*8;
      bf16x16 bfr = make_frag(bp, bp + 16);
      acc[j] = __builtin_amdgcn_wmma_f32_16x16x32_bf16(
          false, a, false, bfr, (short)0, acc[j], false, false);
    }
  }

#pragma unroll
  for (int j = 0; j < 8; ++j) {
    int col = (wave*8 + j)*16 + m;
#pragma unroll
    for (int r = 0; r < 8; ++r) {
      int row = m0 + grp*8 + r;
      out[(size_t)row * E_ + col] = acc[j][r];
    }
  }
}

// ---------------------------------------------------------------------------
extern "C" void kernel_launch(void* const* d_in, const int* in_sizes, int n_in,
                              void* d_out, int out_size, void* d_ws, size_t ws_size,
                              hipStream_t stream) {
  (void)in_sizes; (void)n_in; (void)out_size; (void)ws_size;
  const float* x  = (const float*)d_in[0];
  const float* Wq = (const float*)d_in[1];
  const float* Wf = (const float*)d_in[2];
  const float* Wi = (const float*)d_in[3];
  const float* Wo = (const float*)d_in[4];
  float* out = (float*)d_out;

  char* p = (char*)d_ws;
  auto alloc = [&](size_t bytes) -> char* {
    char* r = p; p += (bytes + 255) & ~(size_t)255; return r;
  };
  float*  qA   = (float*) alloc((size_t)BL_ * F_ * 4);
  float*  kA   = (float*) alloc((size_t)BL_ * F_ * 4);
  float*  vA   = (float*) alloc((size_t)BL_ * F_ * 4);
  float*  gA   = (float*) alloc((size_t)BL_ * F_ * 4);
  float*  aqA  = (float*) alloc((size_t)BL_ * F_ * 4);
  float*  olA  = (float*) alloc((size_t)BL_ * F_ * 4);
  __bf16* obf  = (__bf16*)alloc((size_t)BL_ * F_ * 2);
  __bf16* WT   = (__bf16*)alloc((size_t)N3_ * E_ * 2);
  __bf16* WoT  = (__bf16*)alloc((size_t)E_  * F_ * 2);
  float*  Send = (float*) alloc((size_t)B_ * H_ * NC_ * 4 * 4);
  float*  Gtot = (float*) alloc((size_t)B_ * H_ * NC_ * 2 * 4);
  float*  Sini = (float*) alloc((size_t)B_ * H_ * NC_ * 4 * 4);

  prep_W   <<<(N3_*E_ + 255)/256, 256, 0, stream>>>(Wq, Wf, Wi, WT);
  prep_Wo  <<<(E_*F_  + 255)/256, 256, 0, stream>>>(Wo, WoT);
  proj_gemm<<<BL_/16, 256, 0, stream>>>(x, WT, qA, kA, vA, gA);
  scan_local  <<<B_*NC_, 128, 0, stream>>>(qA, kA, vA, gA, olA, aqA, Send, Gtot);
  scan_combine<<<1, 512, 0, stream>>>(Send, Gtot, Sini);
  scan_fix    <<<B_*NC_, 128, 0, stream>>>(olA, aqA, Sini, obf);
  out_gemm <<<BL_/16, 256, 0, stream>>>(obf, WoT, out);
}
